// MultiHeadAttention_56676388438432
// MI455X (gfx1250) — matrix operand order repaired and checked by reading, not compiled, not run on hardware
//
#include <hip/hip_runtime.h>
#include <stdint.h>

typedef __bf16 bf16_t;
typedef __attribute__((ext_vector_type(16))) __bf16 v16bf;
typedef __attribute__((ext_vector_type(8)))  __bf16 v8bf;
typedef __attribute__((ext_vector_type(8)))  float  v8f;
typedef __attribute__((ext_vector_type(4)))  int    v4i;

#if defined(__gfx1250__) && __has_builtin(__builtin_amdgcn_global_load_async_to_lds_b128)
#define HAVE_ASYNC_LDS 1
#else
#define HAVE_ASYNC_LDS 0
#endif

// ---------- helpers ----------
__device__ __forceinline__ bf16_t f2bf(float f) {
  uint32_t u = __builtin_bit_cast(uint32_t, f);
  uint32_t r = (u + 0x7FFFu + ((u >> 16) & 1u)) >> 16;   // round-to-nearest-even
  return __builtin_bit_cast(bf16_t, (uint16_t)r);
}

// 16-byte global -> LDS copy; async data-mover path on CDNA5, ds_store fallback.
__device__ __forceinline__ void copy16_g2l(const bf16_t* g, bf16_t* l) {
#if HAVE_ASYNC_LDS
  __builtin_amdgcn_global_load_async_to_lds_b128(
      (__attribute__((address_space(1))) v4i*)(uintptr_t)g,
      (__attribute__((address_space(3))) v4i*)(uintptr_t)(__attribute__((address_space(3))) char*)l,
      0, 0);
#else
  *(uint4*)l = *(const uint4*)g;
#endif
}

__device__ __forceinline__ void wait_async_copies() {
#if HAVE_ASYNC_LDS
#if __has_builtin(__builtin_amdgcn_s_wait_asynccnt)
  __builtin_amdgcn_s_wait_asynccnt(0);
#else
  asm volatile("s_wait_asynccnt 0" ::: "memory");
#endif
#endif
}

__device__ __forceinline__ v16bf combine(v8bf lo, v8bf hi) {
  v16bf r;
#pragma unroll
  for (int i = 0; i < 8; ++i) { r[i] = lo[i]; r[8 + i] = hi[i]; }
  return r;
}

// A fragment (global): 16x32 bf16 M x K, ISA 7.12.2 layout.
__device__ __forceinline__ v16bf load_a_frag_g(const bf16_t* A, int lda, int m0, int k0, int lane) {
  int m    = m0 + (lane & 15);
  int koff = k0 + ((lane >> 4) << 3);
  const bf16_t* p = A + (size_t)m * lda + koff;
  return combine(*(const v8bf*)(p), *(const v8bf*)(p + 16));
}

// B fragment (global) from Bt[N][K]: 32x16 bf16 (K x N).
__device__ __forceinline__ v16bf load_b_frag_g(const bf16_t* Bt, int ldb, int n0, int k0, int lane) {
  int n    = n0 + (lane & 15);
  int koff = k0 + ((lane >> 4) << 3);
  const bf16_t* p = Bt + (size_t)n * ldb + koff;
  return combine(*(const v8bf*)(p), *(const v8bf*)(p + 16));
}

// ---------- small prep kernels ----------
__global__ void convert_bf16_kernel(const float* __restrict__ src, bf16_t* __restrict__ dst, size_t n) {
  size_t i = (size_t)blockIdx.x * blockDim.x + threadIdx.x;
  if (i < n) dst[i] = f2bf(src[i]);
}

// dst[c][r] = src[r][c]  (f32 -> bf16), src is R x C row-major
__global__ void transpose_convert_kernel(const float* __restrict__ src, bf16_t* __restrict__ dst, int R, int C) {
  int i = blockIdx.x * blockDim.x + threadIdx.x;
  if (i >= R * C) return;
  int r = i / C, c = i % C;
  dst[(size_t)c * R + r] = f2bf(src[i]);
}

// WoRedT[e][j] = sum_h Wo[h*HD + j][e]   (Wo is D x D row-major, [in][out])
__global__ void reduce_wo_kernel(const float* __restrict__ Wo, bf16_t* __restrict__ WoRedT, int D, int HD) {
  int i = blockIdx.x * blockDim.x + threadIdx.x;
  if (i >= D * HD) return;
  int e = i / HD, j = i % HD;
  float s = 0.f;
#pragma unroll
  for (int h = 0; h < 16; ++h) s += Wo[(size_t)(h * HD + j) * D + e];
  WoRedT[(size_t)e * HD + j] = f2bf(s);
}

// vht[b][j][s] = vh[b][s][j]   (bf16, vh flattened [B*S][HD])
__global__ void transpose_vh_kernel(const bf16_t* __restrict__ vh, bf16_t* __restrict__ vht, int S, int HD) {
  size_t i = (size_t)blockIdx.x * blockDim.x + threadIdx.x;
  int per_b = S * HD;
  int b = (int)(i / per_b);
  int rem = (int)(i % per_b);
  int s = rem / HD, j = rem % HD;
  vht[((size_t)b * HD + j) * S + s] = vh[i];
}

// ---------- LDS double-buffered bf16 WMMA GEMM ----------
// C = alpha * A @ Bt^T + bias ; A: M x K row-major, Bt: N x K row-major.
// Block = 256 thr = 8 waves; block tile 128(M) x 64(N); buffer depth K=64
// (two WMMA sub-steps of 32 per barrier). Async data-mover staging.
#define LDSS 72  // padded LDS row stride in halves (144B -> conflict-free)

__global__ __launch_bounds__(256) void gemm_bf16_wmma(
    const bf16_t* __restrict__ A, const bf16_t* __restrict__ Bt,
    const float* __restrict__ bias,
    float* __restrict__ Cf, bf16_t* __restrict__ Cb,
    int M, int N, int K, float alpha,
    long long sAb, long long sBb, long long sCb) {
  __shared__ bf16_t sA[2][128 * LDSS];
  __shared__ bf16_t sB[2][64 * LDSS];

  int tid  = threadIdx.x;
  int lane = tid & 31;
  int wave = tid >> 5;
  int m0blk = blockIdx.y * 128;
  int n0    = blockIdx.x * 64;
  A  += (size_t)blockIdx.z * sAb;
  Bt += (size_t)blockIdx.z * sBb;

  // cooperative staging of a K=64 slab: each thread issues 6 x 16B async copies
  int srow = tid >> 3;          // 0..31
  int scol = (tid & 7) * 8;     // half offset: 0..56 (16B chunks)
  auto stage = [&](int buf, int k0) {
#pragma unroll
    for (int r = 0; r < 4; ++r)   // A tile: 128 rows
      copy16_g2l(A + (size_t)(m0blk + srow + 32 * r) * K + k0 + scol,
                 &sA[buf][(srow + 32 * r) * LDSS + scol]);
#pragma unroll
    for (int r = 0; r < 2; ++r)   // B tile: 64 rows
      copy16_g2l(Bt + (size_t)(n0 + srow + 32 * r) * K + k0 + scol,
                 &sB[buf][(srow + 32 * r) * LDSS + scol]);
  };

  int a_m    = wave * 16 + (lane & 15);
  int a_koff = (lane >> 4) << 3;
  int b_n    = lane & 15;
  int b_koff = (lane >> 4) << 3;

  stage(0, 0);
  wait_async_copies();
  __syncthreads();

  v8f acc[4] = {};
  int buf = 0;
  for (int k0 = 0; k0 < K; k0 += 64, buf ^= 1) {
    bool more = (k0 + 64) < K;
    if (more) stage(buf ^ 1, k0 + 64);

#pragma unroll
    for (int ks = 0; ks < 64; ks += 32) {
      // batch all fragment loads, then run the WMMA quad back-to-back
      const bf16_t* pa = &sA[buf][a_m * LDSS + ks + a_koff];
      v16bf a = combine(*(const v8bf*)(pa), *(const v8bf*)(pa + 16));
      v16bf bfr[4];
#pragma unroll
      for (int t = 0; t < 4; ++t) {
        const bf16_t* pb = &sB[buf][(t * 16 + b_n) * LDSS + ks + b_koff];
        bfr[t] = combine(*(const v8bf*)(pb), *(const v8bf*)(pb + 16));
      }
#pragma unroll
      for (int t = 0; t < 4; ++t)
        acc[t] = __builtin_amdgcn_wmma_f32_16x16x32_bf16(false, a, false, bfr[t],
                                                         (short)0, acc[t], false, false);
    }
    if (more) {
      wait_async_copies();
      __syncthreads();
    }
  }

  int nl    = lane & 15;
  int mbase = m0blk + wave * 16 + ((lane >> 4) << 3);
  size_t cofs = (size_t)blockIdx.z * sCb;
#pragma unroll
  for (int t = 0; t < 4; ++t) {
    int n = n0 + 16 * t + nl;
    float bs = bias ? bias[n] : 0.f;
#pragma unroll
    for (int r = 0; r < 8; ++r) {
      float v = acc[t][r] * alpha + bs;
      size_t idx = cofs + (size_t)(mbase + r) * N + n;
      if (Cf) Cf[idx] = v;
      if (Cb) Cb[idx] = f2bf(v);
    }
  }
}

// ---------- causal scores: S_tile = (qh @ kh^T) * scale, masked ----------
__global__ __launch_bounds__(256) void scores_kernel(
    const bf16_t* __restrict__ qh, const bf16_t* __restrict__ kh,
    float* __restrict__ scores, int S, float scale) {
  int lane = threadIdx.x & 31;
  int wave = threadIdx.x >> 5;
  int b  = blockIdx.z;
  int m0 = (blockIdx.y * 8 + wave) * 16;
  int n0 = blockIdx.x * 64;
  const bf16_t* Q  = qh + (size_t)b * S * 64;
  const bf16_t* Kh = kh + (size_t)b * S * 64;
  float* Sc = scores + (size_t)b * S * S;

  int nl    = lane & 15;
  int mbase = m0 + ((lane >> 4) << 3);

  if (n0 >= m0 + 16) {  // strip fully above the diagonal (wave-uniform)
#pragma unroll
    for (int t = 0; t < 4; ++t) {
      int n = n0 + 16 * t + nl;
#pragma unroll
      for (int r = 0; r < 8; ++r) Sc[(size_t)(mbase + r) * S + n] = -1e9f;
    }
    return;
  }

  v8f acc[4] = {};
#pragma unroll
  for (int k0 = 0; k0 < 64; k0 += 32) {
    v16bf a = load_a_frag_g(Q, 64, m0, k0, lane);
    v16bf bfr[4];
#pragma unroll
    for (int t = 0; t < 4; ++t)
      bfr[t] = load_b_frag_g(Kh, 64, n0 + 16 * t, k0, lane);
#pragma unroll
    for (int t = 0; t < 4; ++t)
      acc[t] = __builtin_amdgcn_wmma_f32_16x16x32_bf16(false, a, false, bfr[t],
                                                       (short)0, acc[t], false, false);
  }
#pragma unroll
  for (int t = 0; t < 4; ++t) {
    int n = n0 + 16 * t + nl;
#pragma unroll
    for (int r = 0; r < 8; ++r) {
      int m = mbase + r;
      Sc[(size_t)m * S + n] = (n > m) ? -1e9f : acc[t][r] * scale;
    }
  }
}

// ---------- row softmax (f32 in, bf16 out) ----------
__global__ __launch_bounds__(256) void softmax_kernel(
    const float* __restrict__ scores, bf16_t* __restrict__ attn, int S) {
  int row = blockIdx.x, b = blockIdx.y, tid = threadIdx.x;
  const float* src = scores + ((size_t)b * S + row) * S;
  bf16_t*      dst = attn   + ((size_t)b * S + row) * S;
  __shared__ float sm[8];

  float mx = -3.4e38f;
  for (int i = tid; i < S; i += 256) mx = fmaxf(mx, src[i]);
#pragma unroll
  for (int o = 16; o > 0; o >>= 1) mx = fmaxf(mx, __shfl_xor(mx, o, 32));
  if ((tid & 31) == 0) sm[tid >> 5] = mx;
  __syncthreads();
  mx = sm[0];
#pragma unroll
  for (int w = 1; w < 8; ++w) mx = fmaxf(mx, sm[w]);
  __syncthreads();

  float sum = 0.f;
  for (int i = tid; i < S; i += 256) sum += __expf(src[i] - mx);
#pragma unroll
  for (int o = 16; o > 0; o >>= 1) sum += __shfl_xor(sum, o, 32);
  if ((tid & 31) == 0) sm[tid >> 5] = sum;
  __syncthreads();
  sum = 0.f;
#pragma unroll
  for (int w = 0; w < 8; ++w) sum += sm[w];
  float inv = 1.0f / sum;

  for (int i = tid; i < S; i += 256) dst[i] = f2bf(__expf(src[i] - mx) * inv);
}

// ---------- launch ----------
extern "C" void kernel_launch(void* const* d_in, const int* in_sizes, int n_in,
                              void* d_out, int out_size, void* d_ws, size_t ws_size,
                              hipStream_t stream) {
  (void)in_sizes; (void)n_in; (void)out_size; (void)ws_size;
  constexpr int B = 4, S = 2048, D = 1024, HD = 64;
  constexpr size_t MS = (size_t)B * S;   // 8192 rows (batch flattened)

  const float* query = (const float*)d_in[0];
  const float* key   = (const float*)d_in[1];
  const float* value = (const float*)d_in[2];
  /* d_in[3] = mask: causal, applied analytically */
  const float* Wq    = (const float*)d_in[4];
  const float* Wk    = (const float*)d_in[5];
  const float* Wv    = (const float*)d_in[6];
  const float* wq_h  = (const float*)d_in[7];
  const float* bq_h  = (const float*)d_in[8];
  const float* wk_h  = (const float*)d_in[9];
  const float* bk_h  = (const float*)d_in[10];
  const float* wv_h  = (const float*)d_in[11];
  const float* bv_h  = (const float*)d_in[12];
  const float* Wo    = (const float*)d_in[13];

  char* w = (char*)d_ws;
  auto take = [&](size_t bytes) -> char* {
    char* p = w; w += (bytes + 255) & ~(size_t)255; return p;
  };
  bf16_t* Xq    = (bf16_t*)take(MS * D * sizeof(bf16_t));
  bf16_t* Xk    = (bf16_t*)take(MS * D * sizeof(bf16_t));
  bf16_t* Xv    = (bf16_t*)take(MS * D * sizeof(bf16_t));
  bf16_t* qb    = (bf16_t*)take(MS * D * sizeof(bf16_t));
  bf16_t* kb    = (bf16_t*)take(MS * D * sizeof(bf16_t));
  bf16_t* vb    = (bf16_t*)take(MS * D * sizeof(bf16_t));
  bf16_t* WqT   = (bf16_t*)take((size_t)D * D * sizeof(bf16_t));
  bf16_t* WkT   = (bf16_t*)take((size_t)D * D * sizeof(bf16_t));
  bf16_t* WvT   = (bf16_t*)take((size_t)D * D * sizeof(bf16_t));
  bf16_t* wqhT  = (bf16_t*)take((size_t)D * HD * sizeof(bf16_t));
  bf16_t* wkhT  = (bf16_t*)take((size_t)D * HD * sizeof(bf16_t));
  bf16_t* wvhT  = (bf16_t*)take((size_t)D * HD * sizeof(bf16_t));
  bf16_t* WoRT  = (bf16_t*)take((size_t)D * HD * sizeof(bf16_t));
  bf16_t* qhb   = (bf16_t*)take(MS * HD * sizeof(bf16_t));
  bf16_t* khb   = (bf16_t*)take(MS * HD * sizeof(bf16_t));
  bf16_t* vhb   = (bf16_t*)take(MS * HD * sizeof(bf16_t));
  bf16_t* vht   = (bf16_t*)take(MS * HD * sizeof(bf16_t));
  bf16_t* outh  = (bf16_t*)take(MS * HD * sizeof(bf16_t));
  bf16_t* attn  = (bf16_t*)take((size_t)B * S * S * sizeof(bf16_t));
  float*  scrs  = (float*) take((size_t)B * S * S * sizeof(float));

  const dim3 blk(256);
  // 1. convert inputs
  {
    size_t n = MS * D;
    dim3 g((unsigned)((n + 255) / 256));
    convert_bf16_kernel<<<g, blk, 0, stream>>>(query, Xq, n);
    convert_bf16_kernel<<<g, blk, 0, stream>>>(key,   Xk, n);
    convert_bf16_kernel<<<g, blk, 0, stream>>>(value, Xv, n);
  }
  // 2. weight prep
  {
    dim3 g((D * D + 255) / 256);
    transpose_convert_kernel<<<g, blk, 0, stream>>>(Wq, WqT, D, D);
    transpose_convert_kernel<<<g, blk, 0, stream>>>(Wk, WkT, D, D);
    transpose_convert_kernel<<<g, blk, 0, stream>>>(Wv, WvT, D, D);
    dim3 gh((D * HD + 255) / 256);
    transpose_convert_kernel<<<gh, blk, 0, stream>>>(wq_h, wqhT, D, HD);
    transpose_convert_kernel<<<gh, blk, 0, stream>>>(wk_h, wkhT, D, HD);
    transpose_convert_kernel<<<gh, blk, 0, stream>>>(wv_h, wvhT, D, HD);
    reduce_wo_kernel<<<gh, blk, 0, stream>>>(Wo, WoRT, D, HD);
  }
  // 3. full projections: [8192,1024] = [8192,1024] @ [1024,1024]
  {
    dim3 g(D / 64, (unsigned)(MS / 128), 1);
    gemm_bf16_wmma<<<g, blk, 0, stream>>>(Xq, WqT, nullptr, nullptr, qb, (int)MS, D, D, 1.f, 0, 0, 0);
    gemm_bf16_wmma<<<g, blk, 0, stream>>>(Xk, WkT, nullptr, nullptr, kb, (int)MS, D, D, 1.f, 0, 0, 0);
    gemm_bf16_wmma<<<g, blk, 0, stream>>>(Xv, WvT, nullptr, nullptr, vb, (int)MS, D, D, 1.f, 0, 0, 0);
  }
  // 4. head projections: [8192,64] = [8192,1024] @ [1024,64] + bias
  {
    dim3 g(HD / 64, (unsigned)(MS / 128), 1);
    gemm_bf16_wmma<<<g, blk, 0, stream>>>(qb, wqhT, bq_h, nullptr, qhb, (int)MS, HD, D, 1.f, 0, 0, 0);
    gemm_bf16_wmma<<<g, blk, 0, stream>>>(kb, wkhT, bk_h, nullptr, khb, (int)MS, HD, D, 1.f, 0, 0, 0);
    gemm_bf16_wmma<<<g, blk, 0, stream>>>(vb, wvhT, bv_h, nullptr, vhb, (int)MS, HD, D, 1.f, 0, 0, 0);
  }
  // 5. transpose vh per batch -> [B][64][S]
  {
    size_t n = MS * HD;
    transpose_vh_kernel<<<dim3((unsigned)((n + 255) / 256)), blk, 0, stream>>>(vhb, vht, S, HD);
  }
  // 6. causal scores: [B][S][S] f32
  scores_kernel<<<dim3(S / 64, S / 128, B), blk, 0, stream>>>(qhb, khb, scrs, S, 0.125f);
  // 7. softmax -> attn bf16
  softmax_kernel<<<dim3(S, B), blk, 0, stream>>>(scrs, attn, S);
  // 8. out_h = attn @ vh : per batch [2048,64] = [2048,2048] @ [2048,64]
  gemm_bf16_wmma<<<dim3(HD / 64, S / 128, B), blk, 0, stream>>>(
      attn, vht, nullptr, nullptr, outh, S, HD, S, 1.f,
      (long long)S * S, (long long)HD * S, (long long)S * HD);
  // 9. final: d_out = out_h @ Wo_red : [8192,1024] = [8192,64] @ [64,1024]
  gemm_bf16_wmma<<<dim3(D / 64, (unsigned)(MS / 128), 1), blk, 0, stream>>>(
      outh, WoRT, nullptr, (float*)d_out, nullptr, (int)MS, D, HD, 1.f, 0, 0, 0);
}
